// Fastformer_31379031065340
// MI455X (gfx1250) — matrix operand order repaired and checked by reading, not compiled, not run on hardware
//
#include <hip/hip_runtime.h>
#include <hip/hip_bf16.h>

// ---------------------------------------------------------------------------
// Fastformer on MI455X (gfx1250): bf16 WMMA GEMMs with async global->LDS
// double buffering + wave32 softmax reductions
// ---------------------------------------------------------------------------

typedef __attribute__((ext_vector_type(16))) __bf16 v16bf;
typedef __attribute__((ext_vector_type(8)))  __bf16 v8bf;
typedef __attribute__((ext_vector_type(4)))  __bf16 v4bf;
typedef __attribute__((ext_vector_type(8)))  float  v8f;
typedef __attribute__((ext_vector_type(4)))  int    v4i;

#define FF_B 4
#define FF_S 4096
#define FF_D 1024
#define FF_M (FF_B * FF_S)          // 16384 rows for fused-batch GEMMs
#define FF_SCALE 0.03125f           // 1024 ** -0.5

#define TILE_M 128
#define TILE_N 128
#define TILE_K 32
#define RED_ROWS 16                 // rows of s handled per reduction block

// -------- CDNA5 async global->LDS support (guarded; falls back cleanly) ----
#if defined(__AMDGCN__) && __has_builtin(__builtin_amdgcn_global_load_async_to_lds_b128)
#define FF_ASYNC_LDS 1
#endif

#ifdef FF_ASYNC_LDS
__device__ __forceinline__ void ff_async_b128(void* lds_dst, const void* gsrc) {
    __builtin_amdgcn_global_load_async_to_lds_b128(
        (__attribute__((address_space(1))) v4i*)(uintptr_t)gsrc,
        (__attribute__((address_space(3))) v4i*)lds_dst,
        0, 0);
}
#endif

__device__ __forceinline__ void ff_wait_async() {
#ifdef FF_ASYNC_LDS
#if __has_builtin(__builtin_amdgcn_s_wait_asynccnt)
    __builtin_amdgcn_s_wait_asynccnt(0);
#else
    asm volatile("s_wait_asynccnt 0x0" ::: "memory");
#endif
#endif
}

// ------------------------------ small utility kernels ----------------------

__global__ void ff_zero_f32(float* __restrict__ p, int n) {
    int i = blockIdx.x * blockDim.x + threadIdx.x;
    if (i < n) p[i] = 0.0f;
}

// vectorized f32 -> bf16 convert (n4 = n/4 float4 groups)
__global__ void ff_f32_to_bf16(const float* __restrict__ in,
                               __bf16* __restrict__ out, long n4) {
    long i = (long)blockIdx.x * blockDim.x + threadIdx.x;
    if (i < n4) {
        float4 v = ((const float4*)in)[i];
        v4bf o;
        o[0] = (__bf16)v.x; o[1] = (__bf16)v.y;
        o[2] = (__bf16)v.z; o[3] = (__bf16)v.w;
        ((v4bf*)out)[i] = o;
    }
}

// out[z][n][k] = (bf16)( W[k][n] * (useScale ? scale[z][k] : 1) )
// W: [K x N] row-major f32. 32x32 tiles, 256 threads.
__global__ void ff_transpose_scale(const float* __restrict__ W,
                                   const float* __restrict__ scale,
                                   __bf16* __restrict__ out,
                                   int K, int N, int useScale) {
    __shared__ float tile[32][33];
    const int z  = blockIdx.z;
    const int k0 = blockIdx.x * 32;
    const int n0 = blockIdx.y * 32;
    const int tx = threadIdx.x & 31;
    const int ty = threadIdx.x >> 5;     // 0..7
    const float* sc = useScale ? (scale + (long)z * K) : nullptr;

    #pragma unroll
    for (int r = ty; r < 32; r += 8) {
        float v = W[(long)(k0 + r) * N + n0 + tx];
        if (useScale) v *= sc[k0 + r];
        tile[r][tx] = v;
    }
    __syncthreads();
    #pragma unroll
    for (int r = ty; r < 32; r += 8) {
        out[(long)z * K * N + (long)(n0 + r) * K + k0 + tx] = (__bf16)tile[tx][r];
    }
}

// ------------------------------ WMMA GEMM ----------------------------------
// C[M,N](f32 or bf16) = A[M,K](bf16, row-major) * Wt[N,K](bf16, row-major)^T
// MODE 0: store f32   MODE 1: store bf16   MODE 2: add into f32
// Double-buffered LDS tiles; async global->LDS DMA when available.
template <int MODE>
__global__ __launch_bounds__(256)
void ff_gemm_wmma(const __bf16* __restrict__ A,
                  const __bf16* __restrict__ Wt,
                  void* __restrict__ Cout,
                  int M, int N, int Kdim,
                  long aBatchStride, long wBatchStride, long cBatchStride) {
    __shared__ __bf16 ldsA[2][TILE_M][TILE_K + 8];
    __shared__ __bf16 ldsB[2][TILE_N][TILE_K + 8];

    const int bz = blockIdx.z;
    const __bf16* Ab = A  + (long)bz * aBatchStride;
    const __bf16* Wb = Wt + (long)bz * wBatchStride;

    const int m0 = blockIdx.y * TILE_M;
    const int n0 = blockIdx.x * TILE_N;

    const int tid  = threadIdx.x;      // 0..255 (8 waves of 32)
    const int lane = tid & 31;
    const int wave = tid >> 5;
    const int wm = (wave & 3) * 32;    // wave tile origin (M) within block tile
    const int wn = (wave >> 2) * 64;   // wave tile origin (N) within block tile

    v8f acc[2][4] = {};                // 2 x 4 tiles of 16x16 f32

    // Cooperative tile staging: each thread moves one 16-element half-row of
    // the A tile and of the B tile per k-step (2 x b128 each).
    const int lr = tid >> 1;           // 0..127: row within tile
    const int lc = (tid & 1) * 16;     // 0 or 16: k offset
    const __bf16* gaBase = Ab + (long)(m0 + lr) * Kdim + lc;
    const __bf16* gbBase = Wb + (long)(n0 + lr) * Kdim + lc;

    auto stage = [&](int buf, int k0) {
#ifdef FF_ASYNC_LDS
        ff_async_b128(&ldsA[buf][lr][lc],     gaBase + k0);
        ff_async_b128(&ldsA[buf][lr][lc + 8], gaBase + k0 + 8);
        ff_async_b128(&ldsB[buf][lr][lc],     gbBase + k0);
        ff_async_b128(&ldsB[buf][lr][lc + 8], gbBase + k0 + 8);
#else
        *(v8bf*)&ldsA[buf][lr][lc]     = *(const v8bf*)(gaBase + k0);
        *(v8bf*)&ldsA[buf][lr][lc + 8] = *(const v8bf*)(gaBase + k0 + 8);
        *(v8bf*)&ldsB[buf][lr][lc]     = *(const v8bf*)(gbBase + k0);
        *(v8bf*)&ldsB[buf][lr][lc + 8] = *(const v8bf*)(gbBase + k0 + 8);
        __builtin_prefetch(gaBase + k0 + TILE_K, 0, 1);
        __builtin_prefetch(gbBase + k0 + TILE_K, 0, 1);
#endif
    };

    const int lrow   = lane & 15;
    const int khalfA = (lane >> 4) * 8;   // A: lanes<16 take K0-7/K16-23, else K8-15/K24-31
    const int koffB  = (lane >> 4) * 8;   // B: lanes<16 take K0-7/K16-23, else K8-15/K24-31

    // prologue: fill buffer 0
    stage(0, 0);
    ff_wait_async();
    __syncthreads();

    int cur = 0;
    for (int k0 = 0; k0 < Kdim; k0 += TILE_K) {
        // kick off DMA for the next tile into the other buffer while computing
        if (k0 + TILE_K < Kdim) stage(cur ^ 1, k0 + TILE_K);

        v16bf afrag[2], bfrag[4];
        #pragma unroll
        for (int i = 0; i < 2; ++i) {
            const __bf16* p = &ldsA[cur][wm + i * 16 + lrow][0];
            v8bf* h = reinterpret_cast<v8bf*>(&afrag[i]);
            h[0] = *(const v8bf*)(p + khalfA);        // K 0-7   (or 8-15)
            h[1] = *(const v8bf*)(p + 16 + khalfA);   // K 16-23 (or 24-31)
        }
        #pragma unroll
        for (int j = 0; j < 4; ++j) {
            const __bf16* p = &ldsB[cur][wn + j * 16 + lrow][0];
            v8bf* h = reinterpret_cast<v8bf*>(&bfrag[j]);
            h[0] = *(const v8bf*)(p + koffB);         // K 0-7   (or 8-15)
            h[1] = *(const v8bf*)(p + 16 + koffB);    // K 16-23 (or 24-31)
        }
        #pragma unroll
        for (int i = 0; i < 2; ++i)
            #pragma unroll
            for (int j = 0; j < 4; ++j)
                acc[i][j] = __builtin_amdgcn_wmma_f32_16x16x32_bf16(
                    false, afrag[i], false, bfrag[j],
                    (short)0, acc[i][j], false, false);

        ff_wait_async();     // next buffer's DMA (this wave's share) landed
        __syncthreads();     // all waves done reading cur / writing cur^1
        cur ^= 1;
    }

    // C/D layout: VGPR r -> row r (lanes 0-15) and row r+8 (lanes 16-31)
    const int crow = (lane >> 4) * 8;
    const int ccol = lane & 15;
    #pragma unroll
    for (int i = 0; i < 2; ++i)
        #pragma unroll
        for (int j = 0; j < 4; ++j)
            #pragma unroll
            for (int r = 0; r < 8; ++r) {
                const long m = m0 + wm + i * 16 + r + crow;
                const long n = n0 + wn + j * 16 + ccol;
                const float v = acc[i][j][r];
                if (MODE == 0) {
                    ((float*)Cout)[cBatchStride * bz + m * N + n] = v;
                } else if (MODE == 1) {
                    ((__bf16*)Cout)[cBatchStride * bz + m * N + n] = (__bf16)v;
                } else {
                    float* p = (float*)Cout + cBatchStride * bz + m * N + n;
                    *p += v;
                }
            }
}

// ------------------------------ softmax + reduction ------------------------

__device__ __forceinline__ float ff_block_allreduce(float v, bool is_sum) {
    #pragma unroll
    for (int off = 16; off; off >>= 1) {
        float o = __shfl_xor(v, off, 32);
        v = is_sum ? (v + o) : fmaxf(v, o);
    }
    __shared__ float red[8];
    const int wave = threadIdx.x >> 5;
    if ((threadIdx.x & 31) == 0) red[wave] = v;
    __syncthreads();
    float r = red[0];
    #pragma unroll
    for (int w = 1; w < 8; ++w) r = is_sum ? (r + red[w]) : fmaxf(r, red[w]);
    __syncthreads();
    return r;
}

__device__ __forceinline__ void ff_load4(const float* p, float& a, float& b,
                                         float& c, float& d) {
    float4 v = *(const float4*)p;
    a = v.x; b = v.y; c = v.z; d = v.w;
}
__device__ __forceinline__ void ff_load4(const __bf16* p, float& a, float& b,
                                         float& c, float& d) {
    v4bf v = *(const v4bf*)p;
    a = (float)v[0]; b = (float)v[1]; c = (float)v[2]; d = (float)v[3];
}

// v[b,s,:] = (mul ? mul[b,:] : 1) * src[b,s,:]
// alpha = softmax(v * SCALE, axis=-1);  out[b,:] += sum_s v * alpha
template <typename T>
__global__ __launch_bounds__(256)
void ff_reduce(const T* __restrict__ src, const float* __restrict__ mul,
               float* __restrict__ out) {
    const int b = blockIdx.y;
    const int c = threadIdx.x << 2;                       // 4 columns per thread
    const long row0 = (long)b * FF_S + (long)blockIdx.x * RED_ROWS;

    float m0 = 1.f, m1 = 1.f, m2 = 1.f, m3 = 1.f;
    if (mul) {
        const float* mp = mul + (long)b * FF_D + c;
        m0 = mp[0]; m1 = mp[1]; m2 = mp[2]; m3 = mp[3];
    }
    float a0 = 0.f, a1 = 0.f, a2 = 0.f, a3 = 0.f;

    for (int r = 0; r < RED_ROWS; ++r) {
        float v0, v1, v2, v3;
        ff_load4(src + (row0 + r) * FF_D + c, v0, v1, v2, v3);
        v0 *= m0; v1 *= m1; v2 *= m2; v3 *= m3;
        const float t0 = v0 * FF_SCALE, t1 = v1 * FF_SCALE;
        const float t2 = v2 * FF_SCALE, t3 = v3 * FF_SCALE;
        const float mx = ff_block_allreduce(fmaxf(fmaxf(t0, t1), fmaxf(t2, t3)), false);
        const float e0 = __expf(t0 - mx), e1 = __expf(t1 - mx);
        const float e2 = __expf(t2 - mx), e3 = __expf(t3 - mx);
        const float sum = ff_block_allreduce(e0 + e1 + e2 + e3, true);
        const float inv = 1.0f / sum;
        a0 += v0 * e0 * inv; a1 += v1 * e1 * inv;
        a2 += v2 * e2 * inv; a3 += v3 * e3 * inv;
    }
    float* op = out + (long)b * FF_D + c;
    atomicAdd(op + 0, a0); atomicAdd(op + 1, a1);
    atomicAdd(op + 2, a2); atomicAdd(op + 3, a3);
}

// ------------------------------ launch -------------------------------------

// workspace layout (bytes)
#define OFF_XB  0L                                   // x  bf16: 32 MB
#define OFF_KB  (OFF_XB + 33554432L)                 // K  bf16: 32 MB
#define OFF_VB  (OFF_KB + 33554432L)                 // V  bf16: 32 MB
#define OFF_WQT (OFF_VB + 33554432L)                 // Wq^T bf16: 2 MB
#define OFF_WKT (OFF_WQT + 2097152L)
#define OFF_WVT (OFF_WKT + 2097152L)
#define OFF_WRT (OFF_WVT + 2097152L)                 // gk-scaled Wr^T, 4 x 2 MB
#define OFF_GQ  (OFF_WRT + 8388608L)                 // gq: 4x1024 f32
#define OFF_GK  (OFF_GQ + 16384L)                    // gk: 4x1024 f32

extern "C" void kernel_launch(void* const* d_in, const int* in_sizes, int n_in,
                              void* d_out, int out_size, void* d_ws, size_t ws_size,
                              hipStream_t stream) {
    const float* x  = (const float*)d_in[0];
    const float* Wq = (const float*)d_in[1];
    const float* Wk = (const float*)d_in[2];
    const float* Wv = (const float*)d_in[3];
    const float* Wr = (const float*)d_in[4];
    float* out = (float*)d_out;

    char* ws = (char*)d_ws;
    __bf16* XB  = (__bf16*)(ws + OFF_XB);
    __bf16* KB  = (__bf16*)(ws + OFF_KB);
    __bf16* VB  = (__bf16*)(ws + OFF_VB);
    __bf16* WQT = (__bf16*)(ws + OFF_WQT);
    __bf16* WKT = (__bf16*)(ws + OFF_WKT);
    __bf16* WVT = (__bf16*)(ws + OFF_WVT);
    __bf16* WRT = (__bf16*)(ws + OFF_WRT);
    float*  GQ  = (float*)(ws + OFF_GQ);
    float*  GK  = (float*)(ws + OFF_GK);

    // 0) zero gq+gk accumulators (contiguous 8192 floats)
    ff_zero_f32<<<dim3(32), 256, 0, stream>>>(GQ, 2 * FF_B * FF_D);

    // 1) x -> bf16
    const long n4 = (long)FF_M * FF_D / 4;
    ff_f32_to_bf16<<<dim3((unsigned)(n4 / 256)), 256, 0, stream>>>(x, XB, n4);

    // 2) transpose weights to [e][d] bf16
    ff_transpose_scale<<<dim3(32, 32, 1), 256, 0, stream>>>(Wq, nullptr, WQT, FF_D, FF_D, 0);
    ff_transpose_scale<<<dim3(32, 32, 1), 256, 0, stream>>>(Wk, nullptr, WKT, FF_D, FF_D, 0);
    ff_transpose_scale<<<dim3(32, 32, 1), 256, 0, stream>>>(Wv, nullptr, WVT, FF_D, FF_D, 0);

    // 3) projections: Q -> d_out (f32), K -> KB (bf16), V -> VB (bf16)
    const dim3 gqkv(FF_D / TILE_N, FF_M / TILE_M, 1);
    ff_gemm_wmma<0><<<gqkv, 256, 0, stream>>>(XB, WQT, out, FF_M, FF_D, FF_D, 0, 0, 0);
    ff_gemm_wmma<1><<<gqkv, 256, 0, stream>>>(XB, WKT, KB,  FF_M, FF_D, FF_D, 0, 0, 0);
    ff_gemm_wmma<1><<<gqkv, 256, 0, stream>>>(XB, WVT, VB,  FF_M, FF_D, FF_D, 0, 0, 0);

    // 4) gq[b,d] = sum_s q * softmax(q*SCALE)
    ff_reduce<float><<<dim3(FF_S / RED_ROWS, FF_B), 256, 0, stream>>>(out, nullptr, GQ);

    // 5) gk[b,d] = sum_s p * softmax(p*SCALE), p = gq * k
    ff_reduce<__bf16><<<dim3(FF_S / RED_ROWS, FF_B), 256, 0, stream>>>(KB, GQ, GK);

    // 6) per-batch W'[e][d] = gk[b,d] * Wr[d,e]  (folds u = gk ⊙ v into the weight)
    ff_transpose_scale<<<dim3(32, 32, FF_B), 256, 0, stream>>>(Wr, GK, WRT, FF_D, FF_D, 1);

    // 7) out += V @ W'^T  (batched)
    const dim3 gr(FF_D / TILE_N, FF_S / TILE_M, FF_B);
    ff_gemm_wmma<2><<<gr, 256, 0, stream>>>(VB, WRT, out, FF_S, FF_D, FF_D,
                                            (long)FF_S * FF_D, (long)FF_D * FF_D,
                                            (long)FF_S * FF_D);
}
